// MPNNdirected_32890859553364
// MI455X (gfx1250) — matrix operand order repaired and checked by reading, not compiled, not run on hardware
//
#include <hip/hip_runtime.h>

// Problem constants (reference: B=16, F=128, N=2048)
#define BB 16
#define FF 128
#define NN 2048
#define TOTAL (BB * FF * NN)        // 4,194,304 elements of emb_in

typedef __attribute__((ext_vector_type(16))) _Float16 v16h;
typedef __attribute__((ext_vector_type(8)))  float    v8f;
typedef __attribute__((ext_vector_type(4)))  float    v4f;

// ---------------------------------------------------------------------------
// sigmoid: gfx1250 has v_tanh_f32 -> one transcendental instead of exp+rcp.
// ---------------------------------------------------------------------------
__device__ __forceinline__ float fast_sigmoid(float x) {
#if __has_builtin(__builtin_amdgcn_tanhf)
  return __builtin_amdgcn_tanhf(0.5f * x) * 0.5f + 0.5f;   // 0.5*tanh(x/2)+0.5
#else
  float e = __builtin_amdgcn_exp2f(-1.4426950408889634f * x);  // exp(-x)
  return __builtin_amdgcn_rcpf(1.0f + e);
#endif
}

// ---------------------------------------------------------------------------
// Pass 1: per-block partial (sum, sum_sq) over emb_in.  256 blocks x 256 thr,
// v4f grid-stride loads (16 MB read).  Deterministic tree reduction in LDS.
// ---------------------------------------------------------------------------
__global__ __launch_bounds__(256) void reduce_partial_kernel(
    const float* __restrict__ x, float* __restrict__ partials) {
  __shared__ float ssum[256];
  __shared__ float ssq[256];
  const int tid = threadIdx.x;
  const int gid = blockIdx.x * 256 + tid;
  const v4f* __restrict__ x4 = (const v4f*)x;
  float sum = 0.0f, sq = 0.0f;
  const int total4 = TOTAL / 4;
  for (int i = gid; i < total4; i += 256 * 256) {
    v4f t = x4[i];
    sum += (t.x + t.y) + (t.z + t.w);
    sq  += (t.x * t.x + t.y * t.y) + (t.z * t.z + t.w * t.w);
  }
  ssum[tid] = sum;
  ssq[tid]  = sq;
  __syncthreads();
  for (int off = 128; off > 0; off >>= 1) {
    if (tid < off) {
      ssum[tid] += ssum[tid + off];
      ssq[tid]  += ssq[tid + off];
    }
    __syncthreads();
  }
  if (tid == 0) {
    partials[2 * blockIdx.x]     = ssum[0];
    partials[2 * blockIdx.x + 1] = ssq[0];
  }
}

// ---------------------------------------------------------------------------
// Pass 2: reduce 256 partials -> inv_std (torch unbiased std, ddof=1).
// ---------------------------------------------------------------------------
__global__ __launch_bounds__(256) void reduce_final_kernel(
    const float* __restrict__ partials, float* __restrict__ stat) {
  __shared__ float ssum[256];
  __shared__ float ssq[256];
  const int tid = threadIdx.x;
  ssum[tid] = partials[2 * tid];
  ssq[tid]  = partials[2 * tid + 1];
  __syncthreads();
  for (int off = 128; off > 0; off >>= 1) {
    if (tid < off) {
      ssum[tid] += ssum[tid + off];
      ssq[tid]  += ssq[tid + off];
    }
    __syncthreads();
  }
  if (tid == 0) {
    const float cnt  = (float)TOTAL;
    float mean = ssum[0] / cnt;
    float var  = (ssq[0] - cnt * mean * mean) / (cnt - 1.0f);
    stat[0] = 1.0f / sqrtf(var);   // inv_std
  }
}

// ---------------------------------------------------------------------------
// GEMV via WMMA:  s[b, n0+c] = (sum_f v[f] * emb[b,f,n0+c]) * inv_std + bias/2
//
// One wave (32 threads) per 16-column tile.  D = A x B + C with
//   A[m][k] = v[k0+k]            (rows replicated -> every D row equals s)
//   B[k][n] = emb[b][k0+k][n0+n]
// A 16x32 f16 VGPR striping (ISA 7.12.2): lane<16 holds K = {0..7,16..23},
// lane>=16 holds K = {8..15,24..31} of the 32-wide K slice, row = lane&15.
// B 32x16 f16: lane holds column n = lane&15; lanes<16 hold K={0..7,16..23},
// lanes>=16 hold K={8..15,24..31} (2 per VGPR).  K-loop: 4 x wmma_f32_16x16x32_f16.
// D layout: VGPR0, lanes 0..15 = (M=0, N=lane) -> s for this tile.
// ---------------------------------------------------------------------------
__global__ __launch_bounds__(32) void s_wmma_kernel(
    const float* __restrict__ emb, const float* __restrict__ v,
    const float* __restrict__ bias, const float* __restrict__ stat,
    float* __restrict__ s) {
  const int lane = threadIdx.x;        // 0..31
  const int half = lane >> 4;          // 0 or 1
  const int col  = lane & 15;
  const int n0   = blockIdx.x * 16;
  const int b    = blockIdx.y;
  const int n    = n0 + col;
  const float* __restrict__ embB = emb + (size_t)b * FF * NN;

  v8f c = {};
  for (int k0 = 0; k0 < FF; k0 += 32) {
    // A: replicated rows of v (f32 -> f16)
    v16h a;
    const int abase = k0 + half * 8;
#pragma unroll
    for (int e = 0; e < 8; ++e) {
      a[e]     = (_Float16)v[abase + e];
      a[e + 8] = (_Float16)v[abase + 16 + e];
    }
    // B: 32x16 tile of emb, column-per-lane (f32 -> f16)
    v16h bm;
    const int kb = k0 + half * 8;
#pragma unroll
    for (int e = 0; e < 16; ++e)
      bm[e] = (_Float16)embB[(size_t)(kb + e + (e >= 8 ? 8 : 0)) * NN + n];

    c = __builtin_amdgcn_wmma_f32_16x16x32_f16(
        /*neg_a=*/false, a, /*neg_b=*/false, bm,
        /*c_mod=*/(short)0, c, /*reuse_a=*/false, /*reuse_b=*/false);
  }
  if (lane < 16) {
    // fold bias/2 into s so the hot kernel is just s_i + s_j
    s[(size_t)b * NN + n0 + lane] = c[0] * stat[0] + 0.5f * bias[0];
  }
}

// ---------------------------------------------------------------------------
// Hot kernel: adj[b,i,j] = sigmoid(s[b,i] + s[b,j]).  Pure streaming stores.
// Block = 256 threads; tile = 16 rows x 1024 cols.  Per thread: one b128 load
// of 4 s_j values, 16 row scalars in VGPRs, 16 non-temporal b128 stores
// (output is 256 MB > 192 MB L2 and never re-read -> TH=NT).
// ---------------------------------------------------------------------------
__global__ __launch_bounds__(256) void adj_kernel(
    const float* __restrict__ s, float* __restrict__ out) {
  const int tid     = threadIdx.x;
  const int j0      = blockIdx.x * 1024 + tid * 4;   // column start
  const int rowTile = blockIdx.y * 16;               // flat (b*N + i) base
  const int b       = rowTile >> 11;                 // / NN
  const int ibase   = rowTile & (NN - 1);

  const float* __restrict__ srow = s + (size_t)b * NN;
  v4f sj = *(const v4f*)(srow + j0);

  v4f si4[4];
#pragma unroll
  for (int r = 0; r < 4; ++r)
    si4[r] = *(const v4f*)(srow + ibase + r * 4);

  float* outBase = out + (size_t)rowTile * NN + j0;
#pragma unroll
  for (int r = 0; r < 4; ++r) {
#pragma unroll
    for (int c = 0; c < 4; ++c) {
      const float si = si4[r][c];
      v4f y;
      y.x = fast_sigmoid(si + sj.x);
      y.y = fast_sigmoid(si + sj.y);
      y.z = fast_sigmoid(si + sj.z);
      y.w = fast_sigmoid(si + sj.w);
      __builtin_nontemporal_store(y, (v4f*)(outBase + (size_t)(r * 4 + c) * NN));
    }
  }
}

// ---------------------------------------------------------------------------
// Inputs (setup_inputs order): [0]=adj_in (unused), [1]=emb_in [16,128,2048],
// [2]=v [128], [3]=b [1].  Output: adj [16,2048,2048] f32.
// Workspace layout (floats): [0..511] partials, [512] inv_std, [1024..] s.
// ---------------------------------------------------------------------------
extern "C" void kernel_launch(void* const* d_in, const int* in_sizes, int n_in,
                              void* d_out, int out_size, void* d_ws, size_t ws_size,
                              hipStream_t stream) {
  (void)in_sizes; (void)n_in; (void)out_size; (void)ws_size;
  const float* emb  = (const float*)d_in[1];
  const float* v    = (const float*)d_in[2];
  const float* bias = (const float*)d_in[3];
  float* wsf      = (float*)d_ws;
  float* partials = wsf;          // 512 floats
  float* stat     = wsf + 512;    // 1 float (inv_std)
  float* s        = wsf + 1024;   // BB*NN floats = 128 KB

  reduce_partial_kernel<<<256, 256, 0, stream>>>(emb, partials);
  reduce_final_kernel<<<1, 256, 0, stream>>>(partials, stat);
  s_wmma_kernel<<<dim3(NN / 16, BB), 32, 0, stream>>>(emb, v, bias, stat, s);
  adj_kernel<<<dim3(NN / 1024, (BB * NN) / 16), 256, 0, stream>>>(s, (float*)d_out);
}
